// ToMEResidualAttentionBlock_83451214561327
// MI455X (gfx1250) — compile-verified, matrix operand order checked by reading, not run on hardware
//
#include <hip/hip_runtime.h>
#include <math.h>

// ---------------------------------------------------------------------------
// ToME Residual Attention Block for MI455X (gfx1250, wave32, WMMA)
// B=32, N=257, D=1024, H=16, HD=64, R=16
// ---------------------------------------------------------------------------

#define BATCH   32
#define SEQ     257
#define DMODEL  1024
#define NHEAD   16
#define HDIM    64
#define NPAD    288            // SEQ padded to multiple of 32 for flash loop
#define MTOK    (BATCH*SEQ)    // 8224  (divisible by 16)
#define MTOKP   8320           // padded to multiple of 128 (65 tiles)
#define NMERGE  241            // tokens after merging r=16
#define MM      (BATCH*NMERGE) // 7712
#define MMP     7808           // padded to multiple of 128 (61 tiles)
#define RTOP    16
#define NA      129            // even-half tokens
#define NB      128            // odd-half tokens
#define NU      113            // NA - RTOP

typedef __attribute__((ext_vector_type(16))) _Float16 v16h;
typedef __attribute__((ext_vector_type(8)))  _Float16 v8h;
typedef __attribute__((ext_vector_type(8)))  float    v8f;

// ---------------------------------------------------------------------------
// WMMA helpers
// ---------------------------------------------------------------------------
__device__ __forceinline__ v8f wmma_f16(v16h a, v16h b, v8f c) {
  // D = A(16x32 f16) * B(32x16 f16) + C(16x16 f32)
  return __builtin_amdgcn_wmma_f32_16x16x32_f16(
      /*neg_a=*/false, a, /*neg_b=*/false, b,
      /*c_mod=*/(short)0, c, /*reuse_a=*/false, /*reuse_b=*/false);
}

// Load a 16x32 f16 fragment in the CDNA5 A/B lane layout from a row-major
// matrix with leading dimension `ld` (elements). lane l<16: row = row0+l,
// K halves {k0..k0+7, k0+16..k0+23}; lanes>=16 add 8 to the K base.
__device__ __forceinline__ v16h load_frag_g(const _Float16* __restrict__ p,
                                            int row0, int k0, int ld, int lane) {
  const _Float16* base =
      p + (size_t)(row0 + (lane & 15)) * ld + k0 + ((lane >> 4) << 3);
  v8h lo = *(const v8h*)(base);
  v8h hi = *(const v8h*)(base + 16);
  v16h f;
#pragma unroll
  for (int i = 0; i < 8; ++i) { f[i] = lo[i]; f[i + 8] = hi[i]; }
  return f;
}

// ---------------------------------------------------------------------------
// Generic WMMA GEMM:  C[M][N] = A[M][K](f16) * W[N][K](f16)^T  (+ epilogues)
// block = 128 threads (4 waves) in 2x2; each wave owns a 64x64 tile
// (16 WMMA accumulators) -> 2 WMMA per fragment load (~32 FLOP/byte).
// MODE 0: qkv  -> scatter into q16/k16 (head layout) and vT16 (transposed)
// MODE 1: out  -> f32 out + bias + residual
// MODE 2: fc   -> f16 out = gelu(acc + bias)   (exact erf gelu)
// MODE 3: proj -> f32 out + bias + residual (writes d_out)
// ---------------------------------------------------------------------------
template <int MODE>
__global__ __launch_bounds__(128) void gemm_wmma(
    const _Float16* __restrict__ A, const _Float16* __restrict__ W,
    int N, int K, int Mvalid,
    const float* __restrict__ bias,
    const float* __restrict__ resid,
    float* __restrict__ outF,
    _Float16* __restrict__ outH,
    _Float16* __restrict__ q16, _Float16* __restrict__ k16,
    _Float16* __restrict__ vT16) {
  const int lane = threadIdx.x & 31;
  const int wave = threadIdx.x >> 5;
  const int rowBase = blockIdx.y * 128 + (wave >> 1) * 64;
  const int colBase = blockIdx.x * 128 + (wave & 1) * 64;

  v8f acc[4][4] = {};
  for (int k0 = 0; k0 < K; k0 += 32) {
    // B fragments (weight rows = output columns), kept live across the i-loop
    v16h b0 = load_frag_g(W, colBase,      k0, K, lane);
    v16h b1 = load_frag_g(W, colBase + 16, k0, K, lane);
    v16h b2 = load_frag_g(W, colBase + 32, k0, K, lane);
    v16h b3 = load_frag_g(W, colBase + 48, k0, K, lane);
    if (k0 + 32 < K) {
      // CDNA5 speculative prefetch of next k-step cachelines (global_prefetch_b8)
      __builtin_prefetch(A + (size_t)(rowBase + (lane & 15)) * K + k0 + 32, 0, 1);
      __builtin_prefetch(W + (size_t)(colBase + (lane & 15)) * K + k0 + 32, 0, 1);
    }
#pragma unroll
    for (int i = 0; i < 4; ++i) {
      v16h ai = load_frag_g(A, rowBase + 16 * i, k0, K, lane);
      acc[i][0] = wmma_f16(ai, b0, acc[i][0]);
      acc[i][1] = wmma_f16(ai, b1, acc[i][1]);
      acc[i][2] = wmma_f16(ai, b2, acc[i][2]);
      acc[i][3] = wmma_f16(ai, b3, acc[i][3]);
    }
  }

  const int nlo = lane & 15;
  const int mhi = (lane >> 4) * 8;
#pragma unroll
  for (int i = 0; i < 4; ++i) {
#pragma unroll
    for (int j = 0; j < 4; ++j) {
#pragma unroll
      for (int v = 0; v < 8; ++v) {
        int m = rowBase + i * 16 + mhi + v;
        int n = colBase + j * 16 + nlo;
        if (m >= Mvalid) continue;
        float val = acc[i][j][v] + bias[n];
        if constexpr (MODE == 0) {
          int b = m / SEQ, t = m - b * SEQ;
          if (n < DMODEL) {
            int h = n >> 6, d = n & 63;
            q16[(((size_t)(b * NHEAD + h)) * NPAD + t) * HDIM + d] = (_Float16)val;
          } else if (n < 2 * DMODEL) {
            int nn = n - DMODEL; int h = nn >> 6, d = nn & 63;
            k16[(((size_t)(b * NHEAD + h)) * NPAD + t) * HDIM + d] = (_Float16)val;
          } else {
            int nn = n - 2 * DMODEL; int h = nn >> 6, d = nn & 63;
            vT16[(((size_t)(b * NHEAD + h)) * HDIM + d) * NPAD + t] = (_Float16)val;
          }
        } else if constexpr (MODE == 2) {
          float g = 0.5f * val * (1.0f + erff(val * 0.70710678118654752440f));
          outH[(size_t)m * N + n] = (_Float16)g;
        } else {
          outF[(size_t)m * N + n] = val + resid[(size_t)m * N + n];
        }
      }
    }
  }
}

// ---------------------------------------------------------------------------
// Flash attention: 1 wave per 16-row tile per (b,h). Streams 32 columns/step
// with online softmax; O accumulated via WMMA (P staged through LDS to
// convert C-layout f32 -> A-layout f16).
// ---------------------------------------------------------------------------
__global__ __launch_bounds__(32) void attn_kernel(
    const _Float16* __restrict__ q16, const _Float16* __restrict__ k16,
    const _Float16* __restrict__ vT16, _Float16* __restrict__ o16) {
  __shared__ _Float16 sP[16][40];  // 80B rows -> 16B-aligned sub-rows

  const int lane = threadIdx.x & 31;
  const int rt = blockIdx.x;   // row tile 0..16
  const int h  = blockIdx.y;
  const int b  = blockIdx.z;
  const int row0 = rt * 16;

  const _Float16* q  = q16  + ((size_t)(b * NHEAD + h)) * NPAD * HDIM;
  const _Float16* k  = k16  + ((size_t)(b * NHEAD + h)) * NPAD * HDIM;
  const _Float16* vt = vT16 + ((size_t)(b * NHEAD + h)) * HDIM * NPAD;

  const v16h qa0 = load_frag_g(q, row0, 0,  HDIM, lane);
  const v16h qa1 = load_frag_g(q, row0, 32, HDIM, lane);

  v8f o[4] = {};
  float rowMax[8], rowSum[8];
#pragma unroll
  for (int v = 0; v < 8; ++v) { rowMax[v] = -INFINITY; rowSum[v] = 0.0f; }

  const int nlo = lane & 15;
  const int mhi = (lane >> 4) * 8;
  const float scale = 0.125f;  // 1/sqrt(64)

  for (int j0 = 0; j0 < NPAD; j0 += 32) {
    // S tiles: rows [row0,row0+16), cols [j0,j0+16) and [j0+16,j0+32)
    v8f s0 = {}, s1 = {};
    {
      v16h kb;
      kb = load_frag_g(k, j0,      0,  HDIM, lane); s0 = wmma_f16(qa0, kb, s0);
      kb = load_frag_g(k, j0,      32, HDIM, lane); s0 = wmma_f16(qa1, kb, s0);
      kb = load_frag_g(k, j0 + 16, 0,  HDIM, lane); s1 = wmma_f16(qa0, kb, s1);
      kb = load_frag_g(k, j0 + 16, 32, HDIM, lane); s1 = wmma_f16(qa1, kb, s1);
    }
    const bool ok0 = (j0 + nlo) < SEQ;
    const bool ok1 = (j0 + 16 + nlo) < SEQ;
#pragma unroll
    for (int v = 0; v < 8; ++v) {
      s0[v] = ok0 ? s0[v] * scale : -INFINITY;
      s1[v] = ok1 ? s1[v] * scale : -INFINITY;
    }
    // online softmax, per row m = mhi+v (16 lanes of a half-wave share a row)
#pragma unroll
    for (int v = 0; v < 8; ++v) {
      float tmax = fmaxf(s0[v], s1[v]);
#pragma unroll
      for (int sm_ = 8; sm_ >= 1; sm_ >>= 1) tmax = fmaxf(tmax, __shfl_xor(tmax, sm_, 32));
      float nmax = fmaxf(rowMax[v], tmax);
      float alpha = __expf(rowMax[v] - nmax);
      float p0 = __expf(s0[v] - nmax);
      float p1 = __expf(s1[v] - nmax);
      s0[v] = p0; s1[v] = p1;
      float ts = p0 + p1;
#pragma unroll
      for (int sm_ = 8; sm_ >= 1; sm_ >>= 1) ts += __shfl_xor(ts, sm_, 32);
      rowSum[v] = rowSum[v] * alpha + ts;
      rowMax[v] = nmax;
      o[0][v] *= alpha; o[1][v] *= alpha; o[2][v] *= alpha; o[3][v] *= alpha;
    }
    // stage P (16x32 f16) through LDS: C-layout -> memory row-major
#pragma unroll
    for (int v = 0; v < 8; ++v) {
      int m = mhi + v;
      sP[m][nlo]      = (_Float16)s0[v];
      sP[m][16 + nlo] = (_Float16)s1[v];
    }
    asm volatile("s_wait_dscnt 0x0" ::: "memory");  // cross-lane LDS visibility
    // reload P in A-fragment layout
    v16h pf;
    {
      const _Float16* base = &sP[lane & 15][(lane >> 4) << 3];
      v8h lo = *(const v8h*)(base);
      v8h hi = *(const v8h*)(base + 16);
#pragma unroll
      for (int i = 0; i < 8; ++i) { pf[i] = lo[i]; pf[i + 8] = hi[i]; }
    }
    // O(16x64) += P(16x32) @ V(32x64); V columns read as rows of vT
#pragma unroll
    for (int nt = 0; nt < 4; ++nt) {
      v16h vf = load_frag_g(vt, nt * 16, j0, NPAD, lane);
      o[nt] = wmma_f16(pf, vf, o[nt]);
    }
  }

  // normalize + store o in token-major f16 [MTOKP][DMODEL]
#pragma unroll
  for (int v = 0; v < 8; ++v) {
    int m = row0 + mhi + v;
    if (m >= SEQ) continue;
    float inv = 1.0f / rowSum[v];
#pragma unroll
    for (int nt = 0; nt < 4; ++nt) {
      o16[((size_t)(b * SEQ + m)) * DMODEL + h * HDIM + nt * 16 + nlo] =
          (_Float16)(o[nt][v] * inv);
    }
  }
}

// ---------------------------------------------------------------------------
// LayerNorm over 1024-wide rows, f32 in -> f16 out. 128 threads/row.
// ---------------------------------------------------------------------------
__device__ __forceinline__ float blockReduceSum128(float v) {
#pragma unroll
  for (int m = 16; m >= 1; m >>= 1) v += __shfl_xor(v, m, 32);
  __shared__ float sm[4];
  if ((threadIdx.x & 31) == 0) sm[threadIdx.x >> 5] = v;
  __syncthreads();
  float r = sm[0] + sm[1] + sm[2] + sm[3];
  __syncthreads();
  return r;
}

__global__ __launch_bounds__(128) void ln_kernel(
    const float* __restrict__ x, const float* __restrict__ w,
    const float* __restrict__ bb, _Float16* __restrict__ out) {
  const int row = blockIdx.x;
  const float* px = x + (size_t)row * DMODEL;
  float vals[8];
  float s = 0.0f;
#pragma unroll
  for (int i = 0; i < 8; ++i) { vals[i] = px[threadIdx.x + i * 128]; s += vals[i]; }
  s = blockReduceSum128(s);
  const float mu = s * (1.0f / DMODEL);
  float vs = 0.0f;
#pragma unroll
  for (int i = 0; i < 8; ++i) { float d = vals[i] - mu; vs += d * d; }
  vs = blockReduceSum128(vs);
  const float rstd = rsqrtf(vs * (1.0f / DMODEL) + 1e-5f);
#pragma unroll
  for (int i = 0; i < 8; ++i) {
    int c = threadIdx.x + i * 128;
    out[(size_t)row * DMODEL + c] = (_Float16)((vals[i] - mu) * rstd * w[c] + bb[c]);
  }
}

// ---------------------------------------------------------------------------
// Small utility kernels
// ---------------------------------------------------------------------------
__global__ void cvt_f16_kernel(const float* __restrict__ s, _Float16* __restrict__ d, int n) {
  int i = blockIdx.x * 256 + threadIdx.x;
  if (i < n) d[i] = (_Float16)s[i];
}

__global__ void zero_kpad_kernel(_Float16* __restrict__ k16) {
  int i = blockIdx.x * 256 + threadIdx.x;
  const int total = BATCH * NHEAD * (NPAD - SEQ) * HDIM;
  if (i >= total) return;
  int d = i & 63; int r = i >> 6; int tp = r % (NPAD - SEQ); int bh = r / (NPAD - SEQ);
  k16[(((size_t)bh) * NPAD + SEQ + tp) * HDIM + d] = (_Float16)0.0f;
}

__global__ void zero_vpad_kernel(_Float16* __restrict__ vT16) {
  int i = blockIdx.x * 256 + threadIdx.x;
  const int total = BATCH * NHEAD * HDIM * (NPAD - SEQ);
  if (i >= total) return;
  int tp = i % (NPAD - SEQ); int r = i / (NPAD - SEQ); int d = r & 63; int bh = r >> 6;
  vT16[(((size_t)bh) * HDIM + d) * NPAD + SEQ + tp] = (_Float16)0.0f;
}

// metric = mean_h k, then row-normalized. One block (64 thr) per (b,t).
__global__ __launch_bounds__(64) void metric_kernel(
    const _Float16* __restrict__ k16, float* __restrict__ mnorm) {
  const int bt = blockIdx.x;
  const int b = bt / SEQ, t = bt - b * SEQ;
  const int d = threadIdx.x;
  float s = 0.0f;
#pragma unroll
  for (int h = 0; h < NHEAD; ++h)
    s += (float)k16[(((size_t)(b * NHEAD + h)) * NPAD + t) * HDIM + d];
  s *= (1.0f / NHEAD);
  float sq = s * s;
#pragma unroll
  for (int m = 16; m >= 1; m >>= 1) sq += __shfl_xor(sq, m, 32);
  __shared__ float sm[2];
  if ((threadIdx.x & 31) == 0) sm[threadIdx.x >> 5] = sq;
  __syncthreads();
  float norm = sqrtf(sm[0] + sm[1]);
  mnorm[(size_t)bt * HDIM + d] = s / norm;
}

// bipartite scores + row argmax. One block per batch.
__global__ __launch_bounds__(128) void scores_kernel(
    const float* __restrict__ mnorm, float* __restrict__ nmax, int* __restrict__ nidx) {
  const int b = blockIdx.x;
  const float* base = mnorm + (size_t)b * SEQ * HDIM;
  for (int ia = threadIdx.x; ia < NA; ia += 128) {
    if (ia == 0) { nmax[b * NA] = -INFINITY; nidx[b * NA] = 0; continue; }
    const float* a = base + (size_t)(2 * ia) * HDIM;
    float best = -INFINITY; int bi = 0;
    for (int ib = 0; ib < NB; ++ib) {
      const float* bm = base + (size_t)(2 * ib + 1) * HDIM;
      float dot = 0.0f;
#pragma unroll 8
      for (int d = 0; d < HDIM; ++d) dot += a[d] * bm[d];
      if (dot > best) { best = dot; bi = ib; }
    }
    nmax[b * NA + ia] = best;
    nidx[b * NA + ia] = bi;
  }
}

// stable sorts: edge ranking + index tables. One thread per batch (tiny).
__global__ void tables_kernel(const float* __restrict__ nmax, const int* __restrict__ nidx,
                              int* __restrict__ unm, int* __restrict__ src,
                              int* __restrict__ dst) {
  if (threadIdx.x != 0) return;
  const int b = blockIdx.x;
  int ord[NA];
  for (int i = 0; i < NA; ++i) ord[i] = i;
  // stable insertion sort, descending by nmax
  for (int i = 1; i < NA; ++i) {
    int key = ord[i]; float kv = nmax[b * NA + key]; int j = i - 1;
    while (j >= 0 && nmax[b * NA + ord[j]] < kv) { ord[j + 1] = ord[j]; --j; }
    ord[j + 1] = key;
  }
  for (int i = 0; i < RTOP; ++i) {
    src[b * RTOP + i] = ord[i];
    dst[b * RTOP + i] = nidx[b * NA + ord[i]];
  }
  int u[NU];
  for (int i = 0; i < NU; ++i) u[i] = ord[RTOP + i];
  for (int i = 1; i < NU; ++i) {           // ascending insertion sort
    int key = u[i]; int j = i - 1;
    while (j >= 0 && u[j] > key) { u[j + 1] = u[j]; --j; }
    u[j + 1] = key;
  }
  for (int i = 0; i < NU; ++i) unm[b * NU + i] = u[i];
}

// token merge: out row j<NU = unmerged even token; j>=NU = odd token + srcs,
// divided by merge count. grid (NMERGE, BATCH), 128 threads over DMODEL.
__global__ __launch_bounds__(128) void merge_kernel(
    const float* __restrict__ x1, const int* __restrict__ unm,
    const int* __restrict__ src, const int* __restrict__ dst,
    float* __restrict__ xm) {
  const int j = blockIdx.x;
  const int b = blockIdx.y;
  const float* xb = x1 + (size_t)b * SEQ * DMODEL;
  float* out = xm + ((size_t)(b * NMERGE + j)) * DMODEL;
  if (j < NU) {
    const int t = 2 * unm[b * NU + j];
    for (int c = threadIdx.x; c < DMODEL; c += 128)
      out[c] = xb[(size_t)t * DMODEL + c];
  } else {
    const int jd = j - NU;
    const int t = 2 * jd + 1;
    int cnt = 1;
#pragma unroll
    for (int i = 0; i < RTOP; ++i) cnt += (dst[b * RTOP + i] == jd) ? 1 : 0;
    const float inv = 1.0f / (float)cnt;
    for (int c = threadIdx.x; c < DMODEL; c += 128) {
      float s = xb[(size_t)t * DMODEL + c];
#pragma unroll
      for (int i = 0; i < RTOP; ++i)
        if (dst[b * RTOP + i] == jd)
          s += xb[(size_t)(2 * src[b * RTOP + i]) * DMODEL + c];
      out[c] = s * inv;
    }
  }
}

// ---------------------------------------------------------------------------
// Host-side orchestration
// ---------------------------------------------------------------------------
static inline size_t alignup256(size_t x) { return (x + 255) & ~(size_t)255; }

extern "C" void kernel_launch(void* const* d_in, const int* in_sizes, int n_in,
                              void* d_out, int out_size, void* d_ws, size_t ws_size,
                              hipStream_t stream) {
  (void)in_sizes; (void)n_in; (void)out_size; (void)ws_size;
  const float* q_x       = (const float*)d_in[0];
  const float* ln1_w     = (const float*)d_in[1];
  const float* ln1_b     = (const float*)d_in[2];
  const float* in_proj_w = (const float*)d_in[3];
  const float* in_proj_b = (const float*)d_in[4];
  const float* out_w     = (const float*)d_in[5];
  const float* out_b     = (const float*)d_in[6];
  const float* ln2_w     = (const float*)d_in[7];
  const float* ln2_b     = (const float*)d_in[8];
  const float* fc_w      = (const float*)d_in[9];
  const float* fc_b      = (const float*)d_in[10];
  const float* proj_w    = (const float*)d_in[11];
  const float* proj_b    = (const float*)d_in[12];
  float* outp = (float*)d_out;

  char* w = (char*)d_ws;
  size_t off = 0;
  auto alloc = [&](size_t bytes) { void* p = w + off; off = alignup256(off + bytes); return p; };

  _Float16* h16     = (_Float16*)alloc((size_t)MTOKP * DMODEL * 2);
  _Float16* wqkv16  = (_Float16*)alloc((size_t)3 * DMODEL * DMODEL * 2);
  _Float16* wout16  = (_Float16*)alloc((size_t)DMODEL * DMODEL * 2);
  _Float16* wfc16   = (_Float16*)alloc((size_t)4 * DMODEL * DMODEL * 2);
  _Float16* wproj16 = (_Float16*)alloc((size_t)DMODEL * 4 * DMODEL * 2);
  _Float16* q16     = (_Float16*)alloc((size_t)BATCH * NHEAD * NPAD * HDIM * 2);
  _Float16* k16     = (_Float16*)alloc((size_t)BATCH * NHEAD * NPAD * HDIM * 2);
  _Float16* vT16    = (_Float16*)alloc((size_t)BATCH * NHEAD * HDIM * NPAD * 2);
  _Float16* o16     = (_Float16*)alloc((size_t)MTOKP * DMODEL * 2);
  float*    x1      = (float*)   alloc((size_t)MTOKP * DMODEL * 4);
  float*    mnorm   = (float*)   alloc((size_t)BATCH * SEQ * HDIM * 4);
  float*    nmax    = (float*)   alloc((size_t)BATCH * NA * 4);
  int*      nidx    = (int*)     alloc((size_t)BATCH * NA * 4);
  int*      unm     = (int*)     alloc((size_t)BATCH * NU * 4);
  int*      srci    = (int*)     alloc((size_t)BATCH * RTOP * 4);
  int*      dsti    = (int*)     alloc((size_t)BATCH * RTOP * 4);
  float*    xm      = (float*)   alloc((size_t)MMP * DMODEL * 4);
  _Float16* h2      = (_Float16*)alloc((size_t)MMP * DMODEL * 2);
  _Float16* g16     = (_Float16*)alloc((size_t)MMP * 4 * DMODEL * 2);

  // 1) convert weights to f16
  auto cvt = [&](const float* s, _Float16* d, int n) {
    cvt_f16_kernel<<<(n + 255) / 256, 256, 0, stream>>>(s, d, n);
  };
  cvt(in_proj_w, wqkv16,  3 * DMODEL * DMODEL);
  cvt(out_w,     wout16,  DMODEL * DMODEL);
  cvt(fc_w,      wfc16,   4 * DMODEL * DMODEL);
  cvt(proj_w,    wproj16, DMODEL * 4 * DMODEL);

  // 2) zero padded token regions of k / vT (masked softmax keeps them inert)
  {
    int nk = BATCH * NHEAD * (NPAD - SEQ) * HDIM;
    zero_kpad_kernel<<<(nk + 255) / 256, 256, 0, stream>>>(k16);
    zero_vpad_kernel<<<(nk + 255) / 256, 256, 0, stream>>>(vT16);
  }

  // 3) LN1 -> f16 activations
  ln_kernel<<<MTOK, 128, 0, stream>>>(q_x, ln1_w, ln1_b, h16);

  // 4) qkv GEMM (WMMA), scatter into head-major q/k and transposed v
  gemm_wmma<0><<<dim3(3 * DMODEL / 128, MTOKP / 128), 128, 0, stream>>>(
      h16, wqkv16, 3 * DMODEL, DMODEL, MTOK, in_proj_b,
      nullptr, nullptr, nullptr, q16, k16, vT16);

  // 5) flash attention (WMMA QK^T + online softmax + WMMA PV)
  attn_kernel<<<dim3((SEQ + 15) / 16, NHEAD, BATCH), 32, 0, stream>>>(q16, k16, vT16, o16);

  // 6) out projection + residual -> x1 (f32)
  gemm_wmma<1><<<dim3(DMODEL / 128, MTOKP / 128), 128, 0, stream>>>(
      o16, wout16, DMODEL, DMODEL, MTOK, out_b,
      q_x, x1, nullptr, nullptr, nullptr, nullptr);

  // 7) ToME: metric -> scores/argmax -> rank tables -> merge
  metric_kernel<<<BATCH * SEQ, 64, 0, stream>>>(k16, mnorm);
  scores_kernel<<<BATCH, 128, 0, stream>>>(mnorm, nmax, nidx);
  tables_kernel<<<BATCH, 32, 0, stream>>>(nmax, nidx, unm, srci, dsti);
  merge_kernel<<<dim3(NMERGE, BATCH), 128, 0, stream>>>(x1, unm, srci, dsti, xm);

  // 8) LN2 -> f16
  ln_kernel<<<MM, 128, 0, stream>>>(xm, ln2_w, ln2_b, h2);

  // 9) MLP: fc + exact GELU (WMMA), then proj + residual (WMMA) -> d_out
  gemm_wmma<2><<<dim3(4 * DMODEL / 128, MMP / 128), 128, 0, stream>>>(
      h2, wfc16, 4 * DMODEL, DMODEL, MM, fc_b,
      nullptr, nullptr, g16, nullptr, nullptr, nullptr);
  gemm_wmma<3><<<dim3(DMODEL / 128, MMP / 128), 128, 0, stream>>>(
      g16, wproj16, DMODEL, 4 * DMODEL, MM, proj_b,
      xm, outp, nullptr, nullptr, nullptr, nullptr);
}
